// CaptionDecoder_73280732005170
// MI455X (gfx1250) — matrix operand order repaired and checked by reading, not compiled, not run on hardware
//
#include <hip/hip_runtime.h>
#include <cstdint>
#include <cstddef>

// Problem constants
#define B_  64
#define TC  21
#define TS  20      // decode steps = Tc-1
#define P_  196
#define E_  512
#define H_  512
#define V_  30000

// ---------------------------------------------------------------------------
// WMMA types (CDNA5 gfx1250, wave32): V_WMMA_F32_16X16X32_BF16
// ---------------------------------------------------------------------------
typedef __attribute__((ext_vector_type(16))) __bf16        v16bf;
typedef __attribute__((ext_vector_type(8)))  float         v8f;
typedef __attribute__((ext_vector_type(8)))  unsigned short us8;

union V16U { v16bf v; us8 h[2]; };

__device__ __forceinline__ unsigned short f2bf(float f) {
  union { float f; unsigned u; } v; v.f = f;
  unsigned r = v.u + 0x7fffu + ((v.u >> 16) & 1u);   // round-to-nearest-even
  return (unsigned short)(r >> 16);
}
__device__ __forceinline__ float sigmoidf(float x) { return 1.0f / (1.0f + __expf(-x)); }

// Fragment loaders (per ISA 16-bit A layout and dense-B layout):
//   A: row m = lane&15; lane<16 -> K {0..7,16..23}, lane>=16 -> K {8..15,24..31}
//   B: col n = lane&15; lane<16 -> K {0..7,16..23}, lane>=16 -> K {8..15,24..31}
__device__ __forceinline__ v16bf load_fragA(const unsigned short* Arow, int k0) {
  V16U a;
  a.h[0] = *reinterpret_cast<const us8*>(Arow + k0);
  a.h[1] = *reinterpret_cast<const us8*>(Arow + k0 + 16);
  return a.v;
}
__device__ __forceinline__ v16bf load_fragB(const unsigned short* Brow, int k0) {
  V16U b;
  b.h[0] = *reinterpret_cast<const us8*>(Brow + k0);
  b.h[1] = *reinterpret_cast<const us8*>(Brow + k0 + 16);
  return b.v;
}

// ---------------------------------------------------------------------------
// Small-GEMM kernel (latency-bound serial chain): one wave -> 16x16 tile.
// Maximizes wave parallelism for M=64, modest-N GEMMs.
// ---------------------------------------------------------------------------
__global__ __launch_bounds__(256) void gemm_bf16_wmma_16(
    const unsigned short* __restrict__ A,   // [M][K] bf16 bits
    const unsigned short* __restrict__ Bt,  // [N][K] bf16 bits (pre-transposed)
    const float* __restrict__ bias,         // [N] or nullptr
    float* __restrict__ C,                  // [M][ldc] fp32
    int M, int N, int K, int ldc)
{
  const int lane = threadIdx.x & 31;
  const int wid  = (blockIdx.x * blockDim.x + threadIdx.x) >> 5;
  const int mT = M >> 4, nT = N >> 4;
  if (wid >= mT * nT) return;               // wave-uniform exit; EXEC stays all-1s
  const int mt  = wid % mT;
  const int nt  = wid / mT;
  const int hi  = lane >> 4;
  const int l15 = lane & 15;

  const unsigned short* Arow = A  + (size_t)(mt * 16 + l15) * K + hi * 8;
  const unsigned short* Brow = Bt + (size_t)(nt * 16 + l15) * K + hi * 8;

  v8f acc = {};
  for (int k0 = 0; k0 < K; k0 += 32) {
    acc = __builtin_amdgcn_wmma_f32_16x16x32_bf16(
        false, load_fragA(Arow, k0), false, load_fragB(Brow, k0),
        (short)0, acc, false, false);
  }

  const int   n  = nt * 16 + l15;
  const float bv = bias ? bias[n] : 0.0f;
  const int   m0 = mt * 16 + hi * 8;        // C/D: VGPR r -> M = r + 8*(lane>=16)
#pragma unroll
  for (int r = 0; r < 8; ++r)
    C[(size_t)(m0 + r) * ldc + n] = acc[r] + bv;
}

// ---------------------------------------------------------------------------
// Streaming-GEMM kernel (L2-bandwidth-bound: logits / enc_proj):
// one wave -> 64x16 tile via 4 accumulators; the weight (B) fragment is
// loaded ONCE per K-step and reused by 4 WMMAs -> weights stream read exactly
// once, load:wmma ratio drops from 4:1 to 2.5:1. Requires M % 64 == 0.
// ---------------------------------------------------------------------------
__global__ __launch_bounds__(256) void gemm_bf16_wmma_64(
    const unsigned short* __restrict__ A,   // [M][K] bf16 bits
    const unsigned short* __restrict__ Bt,  // [N][K] bf16 bits (pre-transposed)
    const float* __restrict__ bias,         // [N] or nullptr
    float* __restrict__ C,                  // [M][ldc] fp32
    int M, int N, int K, int ldc)
{
  const int lane = threadIdx.x & 31;
  const int wid  = (blockIdx.x * blockDim.x + threadIdx.x) >> 5;
  const int mT4 = M >> 6, nT = N >> 4;
  if (wid >= mT4 * nT) return;              // wave-uniform exit
  const int mt  = wid % mT4;                // 64-row super-tile
  const int nt  = wid / mT4;
  const int hi  = lane >> 4;
  const int l15 = lane & 15;

  const unsigned short* Arow0 = A + (size_t)(mt * 64 + l15) * K + hi * 8;
  const size_t rowStride16 = (size_t)16 * K;
  const unsigned short* Brow = Bt + (size_t)(nt * 16 + l15) * K + hi * 8;

  v8f acc0 = {}, acc1 = {}, acc2 = {}, acc3 = {};
  for (int k0 = 0; k0 < K; k0 += 32) {
    const v16bf bfrag = load_fragB(Brow, k0);           // loaded once, used 4x
    acc0 = __builtin_amdgcn_wmma_f32_16x16x32_bf16(
        false, load_fragA(Arow0,                 k0), false, bfrag, (short)0, acc0, false, false);
    acc1 = __builtin_amdgcn_wmma_f32_16x16x32_bf16(
        false, load_fragA(Arow0 + rowStride16,   k0), false, bfrag, (short)0, acc1, false, false);
    acc2 = __builtin_amdgcn_wmma_f32_16x16x32_bf16(
        false, load_fragA(Arow0 + 2*rowStride16, k0), false, bfrag, (short)0, acc2, false, false);
    acc3 = __builtin_amdgcn_wmma_f32_16x16x32_bf16(
        false, load_fragA(Arow0 + 3*rowStride16, k0), false, bfrag, (short)0, acc3, false, false);
  }

  const int   n  = nt * 16 + l15;
  const float bv = bias ? bias[n] : 0.0f;
  const int   mbase = mt * 64 + hi * 8;
  float* __restrict__ Cp = C + n;
#pragma unroll
  for (int r = 0; r < 8; ++r) Cp[(size_t)(mbase      + r) * ldc] = acc0[r] + bv;
#pragma unroll
  for (int r = 0; r < 8; ++r) Cp[(size_t)(mbase + 16 + r) * ldc] = acc1[r] + bv;
#pragma unroll
  for (int r = 0; r < 8; ++r) Cp[(size_t)(mbase + 32 + r) * ldc] = acc2[r] + bv;
#pragma unroll
  for (int r = 0; r < 8; ++r) Cp[(size_t)(mbase + 48 + r) * ldc] = acc3[r] + bv;
}

// ---------------------------------------------------------------------------
// Weight transpose + fp32 -> bf16:  Wt[n*K + k] = bf16(W[k*N + n])
// ---------------------------------------------------------------------------
__global__ __launch_bounds__(256) void transpose_f32_bf16(
    const float* __restrict__ W, unsigned short* __restrict__ Wt, int K, int N)
{
  size_t idx = (size_t)blockIdx.x * 256 + threadIdx.x;
  size_t total = (size_t)K * N;
  if (idx >= total) return;
  size_t n = idx / (size_t)K, k = idx % (size_t)K;
  Wt[idx] = f2bf(W[k * (size_t)N + n]);
}

__global__ __launch_bounds__(256) void f32_to_bf16_k(
    const float* __restrict__ in, unsigned short* __restrict__ out, size_t n)
{
  size_t i = (size_t)blockIdx.x * 256 + threadIdx.x;
  if (i < n) out[i] = f2bf(in[i]);
}

// hb = ha(=h0), cb = ca(=c0), hb_bf = bf16(h0)
__global__ __launch_bounds__(256) void init_state_k(
    const float* __restrict__ ha, const float* __restrict__ ca,
    float* __restrict__ hb, float* __restrict__ cb, unsigned short* __restrict__ hb_bf)
{
  int i = blockIdx.x * 256 + threadIdx.x;   // 64*512 total
  float h = ha[i];
  hb[i] = h; cb[i] = ca[i]; hb_bf[i] = f2bf(h);
}

// ---------------------------------------------------------------------------
// Attention: scores = tanh(enc_proj + dec_proj)·v_w + v_b; softmax over P;
// context = alpha·encoder_out; then builds xcat0 = bf16([embed | context | ha]).
// One block per batch item; wave32 shuffle reductions.
// ---------------------------------------------------------------------------
__global__ __launch_bounds__(256) void attention_k(
    const float* __restrict__ dec_proj,  // [B][E]
    const float* __restrict__ enc_proj,  // [B][P][E]
    const float* __restrict__ enc,       // encoder_out [B][P][E]
    const float* __restrict__ v_w,       // [E]
    const float* __restrict__ v_b,       // [1]
    const float* __restrict__ emb,       // [V][E]
    const int*   __restrict__ captions,  // [B][TC]
    const float* __restrict__ ha,        // [B][E] (carry, pre-LSTM0)
    unsigned short* __restrict__ xcat0,  // [B][3E] bf16
    int t)
{
  __shared__ float s_dec[E_];
  __shared__ float s_sc[P_];
  const int b = blockIdx.x;
  const int tid = threadIdx.x;
  const int wave = tid >> 5, lane = tid & 31;

  for (int e = tid; e < E_; e += 256) s_dec[e] = dec_proj[b * E_ + e];
  __syncthreads();

  // scores: one wave per attention position
  for (int p = wave; p < P_; p += 8) {
    const float* er = enc_proj + ((size_t)b * P_ + p) * E_;
    float acc = 0.f;
    for (int e = lane; e < E_; e += 32)
      acc += tanhf(er[e] + s_dec[e]) * v_w[e];
#pragma unroll
    for (int off = 16; off > 0; off >>= 1) acc += __shfl_down(acc, off, 32);
    if (lane == 0) s_sc[p] = acc + v_b[0];
  }
  __syncthreads();

  // softmax over P (wave 0)
  if (wave == 0) {
    float mx = -3.4e38f;
    for (int p = lane; p < P_; p += 32) mx = fmaxf(mx, s_sc[p]);
#pragma unroll
    for (int off = 16; off > 0; off >>= 1) mx = fmaxf(mx, __shfl_xor(mx, off, 32));
    float sum = 0.f;
    for (int p = lane; p < P_; p += 32) { float ex = __expf(s_sc[p] - mx); s_sc[p] = ex; sum += ex; }
#pragma unroll
    for (int off = 16; off > 0; off >>= 1) sum += __shfl_xor(sum, off, 32);
    float inv = 1.0f / sum;
    for (int p = lane; p < P_; p += 32) s_sc[p] *= inv;
  }
  __syncthreads();

  // context + assemble xcat0 = [embed | context | ha] in bf16
  const int tok = captions[b * TC + t];
  unsigned short* xr = xcat0 + (size_t)b * (3 * E_);
  for (int e = tid; e < E_; e += 256) {
    float ctx = 0.f;
    const float* ep = enc + (size_t)b * P_ * E_ + e;
    for (int p = 0; p < P_; ++p) ctx += s_sc[p] * ep[(size_t)p * E_];
    xr[e]            = f2bf(emb[(size_t)tok * E_ + e]);
    xr[E_ + e]       = f2bf(ctx);
    xr[2 * E_ + e]   = f2bf(ha[b * E_ + e]);
  }
}

// ---------------------------------------------------------------------------
// LSTM cell epilogue: g = [f|i|ct|o] pre-activations (bias already added by GEMM)
// Optionally emits xcat_next = bf16([h_new | h_other_old]) and h_bf = bf16(h_new).
// ---------------------------------------------------------------------------
__global__ __launch_bounds__(256) void lstm_k(
    const float* __restrict__ g,               // [B][4H]
    float* __restrict__ h, float* __restrict__ c,  // [B][H] states, in-place
    unsigned short* __restrict__ xcat_next,    // [B][2H] bf16 or nullptr
    const float* __restrict__ h_old_other,     // [B][H] (other cell's old h) or nullptr
    unsigned short* __restrict__ h_bf)         // [B][H] bf16 or nullptr
{
  const int idx = blockIdx.x * 256 + threadIdx.x;  // 0 .. B*H-1
  const int b = idx >> 9, j = idx & 511;
  const float* gr = g + (size_t)b * (4 * H_);
  const float f  = sigmoidf(gr[j]);
  const float in = sigmoidf(gr[H_ + j]);
  const float ct = tanhf(gr[2 * H_ + j]);
  const float o  = sigmoidf(gr[3 * H_ + j]);
  const float cv = f * c[idx] + in * ct;
  const float hv = o * tanhf(cv);
  c[idx] = cv;
  h[idx] = hv;
  if (xcat_next) {
    unsigned short* xr = xcat_next + (size_t)b * (2 * H_);
    xr[j]       = f2bf(hv);
    xr[H_ + j]  = f2bf(h_old_other[idx]);
  }
  if (h_bf) h_bf[idx] = f2bf(hv);
}

// ---------------------------------------------------------------------------
// Host-side orchestration
// ---------------------------------------------------------------------------
extern "C" void kernel_launch(void* const* d_in, const int* in_sizes, int n_in,
                              void* d_out, int out_size, void* d_ws, size_t ws_size,
                              hipStream_t stream) {
  (void)in_sizes; (void)n_in; (void)out_size; (void)ws_size;

  const float* features = (const float*)d_in[0];
  const float* enc      = (const float*)d_in[1];   // encoder_out
  const float* emb      = (const float*)d_in[2];
  const float* W_enc    = (const float*)d_in[3];
  const float* b_enc    = (const float*)d_in[4];
  const float* W_dec    = (const float*)d_in[5];
  const float* b_dec    = (const float*)d_in[6];
  const float* v_w      = (const float*)d_in[7];
  const float* v_b      = (const float*)d_in[8];
  const float* W_g0     = (const float*)d_in[9];
  const float* b_g0     = (const float*)d_in[10];
  const float* W_g1     = (const float*)d_in[11];
  const float* b_g1     = (const float*)d_in[12];
  const float* W_out    = (const float*)d_in[13];
  const float* b_out    = (const float*)d_in[14];
  const float* W_ih     = (const float*)d_in[15];
  const float* b_ih     = (const float*)d_in[16];
  const float* W_ic     = (const float*)d_in[17];
  const float* b_ic     = (const float*)d_in[18];
  const int*   captions = (const int*)d_in[19];
  float* out = (float*)d_out;

  // Workspace carve-out (256B aligned)
  size_t off = 0;
  auto take  = [&](size_t bytes) -> void* {
    void* p = (void*)((char*)d_ws + off);
    off = (off + bytes + 255) & ~(size_t)255;
    return p;
  };
  auto takeU = [&](size_t elems) { return (unsigned short*)take(elems * 2); };
  auto takeF = [&](size_t elems) { return (float*)take(elems * 4); };

  unsigned short* wt_enc = takeU((size_t)E_ * E_);            // [A=512][K=512]
  unsigned short* wt_dec = takeU((size_t)E_ * E_);
  unsigned short* wt_ih  = takeU((size_t)E_ * E_);
  unsigned short* wt_ic  = takeU((size_t)E_ * E_);
  unsigned short* wt_g0  = takeU((size_t)(4 * H_) * (3 * E_)); // [2048][1536]
  unsigned short* wt_g1  = takeU((size_t)(4 * H_) * (2 * H_)); // [2048][1024]
  unsigned short* wt_out = takeU((size_t)V_ * H_);             // [30000][512]
  unsigned short* feat_bf = takeU((size_t)B_ * E_);
  unsigned short* enc_bf  = takeU((size_t)B_ * P_ * E_);
  float* enc_proj = takeF((size_t)B_ * P_ * E_);
  float* ha = takeF((size_t)B_ * H_);
  float* ca = takeF((size_t)B_ * H_);
  float* hb = takeF((size_t)B_ * H_);
  float* cb = takeF((size_t)B_ * H_);
  unsigned short* hb_bf = takeU((size_t)B_ * H_);
  float* dec_proj = takeF((size_t)B_ * E_);
  unsigned short* xcat0 = takeU((size_t)B_ * (3 * E_));
  unsigned short* xcat1 = takeU((size_t)B_ * (2 * H_));
  float* g0 = takeF((size_t)B_ * 4 * H_);
  float* g1 = takeF((size_t)B_ * 4 * H_);

  auto transpose = [&](const float* W, unsigned short* Wt, int K, int N) {
    size_t total = (size_t)K * N;
    int blocks = (int)((total + 255) / 256);
    transpose_f32_bf16<<<blocks, 256, 0, stream>>>(W, Wt, K, N);
  };
  // Small/serial GEMM: 16x16 tiles, max wave parallelism
  auto gemm16 = [&](const unsigned short* A, const unsigned short* Bt, const float* bias,
                    float* C, int M, int N, int K, int ldc) {
    int tiles = (M / 16) * (N / 16);
    int blocks = (tiles + 7) / 8;           // 8 waves per 256-thread block
    gemm_bf16_wmma_16<<<blocks, 256, 0, stream>>>(A, Bt, bias, C, M, N, K, ldc);
  };
  // Streaming GEMM: 64x16 tiles, weight stream read once (M % 64 == 0)
  auto gemm64 = [&](const unsigned short* A, const unsigned short* Bt, const float* bias,
                    float* C, int M, int N, int K, int ldc) {
    int tiles = (M / 64) * (N / 16);
    int blocks = (tiles + 7) / 8;
    gemm_bf16_wmma_64<<<blocks, 256, 0, stream>>>(A, Bt, bias, C, M, N, K, ldc);
  };

  // ---- One-time (per launch) weight prep: fp32 -> bf16 transposed [N][K]
  transpose(W_enc, wt_enc, E_, E_);
  transpose(W_dec, wt_dec, E_, E_);
  transpose(W_ih,  wt_ih,  E_, E_);
  transpose(W_ic,  wt_ic,  E_, E_);
  transpose(W_g0,  wt_g0,  3 * E_, 4 * H_);
  transpose(W_g1,  wt_g1,  2 * H_, 4 * H_);
  transpose(W_out, wt_out, H_, V_);

  // Activation conversions
  {
    size_t nf = (size_t)B_ * E_;
    f32_to_bf16_k<<<(int)((nf + 255) / 256), 256, 0, stream>>>(features, feat_bf, nf);
    size_t ne = (size_t)B_ * P_ * E_;
    f32_to_bf16_k<<<(int)((ne + 255) / 256), 256, 0, stream>>>(enc, enc_bf, ne);
  }

  // ---- Initial states: h0 = features@W_ih + b_ih, c0 = features@W_ic + b_ic
  gemm16(feat_bf, wt_ih, b_ih, ha, B_, H_, E_, H_);
  gemm16(feat_bf, wt_ic, b_ic, ca, B_, H_, E_, H_);
  init_state_k<<<(B_ * H_) / 256, 256, 0, stream>>>(ha, ca, hb, cb, hb_bf);

  // ---- enc_proj = encoder_out @ W_enc + b_enc  (M = B*P = 12544, streaming)
  gemm64(enc_bf, wt_enc, b_enc, enc_proj, B_ * P_, E_, E_, E_);

  // ---- Sequential decode: 20 steps, 7 kernels each (stream ordering = deps)
  for (int t = 0; t < TS; ++t) {
    gemm16(hb_bf, wt_dec, b_dec, dec_proj, B_, E_, E_, E_);
    attention_k<<<B_, 256, 0, stream>>>(dec_proj, enc_proj, enc, v_w, v_b, emb,
                                        captions, ha, xcat0, t);
    gemm16(xcat0, wt_g0, b_g0, g0, B_, 4 * H_, 3 * E_, 4 * H_);
    lstm_k<<<(B_ * H_) / 256, 256, 0, stream>>>(g0, ha, ca, xcat1, hb,
                                                (unsigned short*)nullptr);
    gemm16(xcat1, wt_g1, b_g1, g1, B_, 4 * H_, 2 * H_, 4 * H_);
    lstm_k<<<(B_ * H_) / 256, 256, 0, stream>>>(g1, hb, cb, (unsigned short*)nullptr,
                                                (const float*)nullptr, hb_bf);
    // logits[b][t][v]: C base = out + t*V, row stride = TS*V (streaming GEMM)
    gemm64(hb_bf, wt_out, b_out, out + (size_t)t * V_, B_, V_, H_, TS * V_);
  }
}
